// CNNMetaAttention_11347303596210
// MI455X (gfx1250) — matrix operand order repaired and checked by reading, not compiled, not run on hardware
//
#include <hip/hip_runtime.h>

typedef __bf16 bf16;
typedef __attribute__((ext_vector_type(16))) __bf16 v16bf;
typedef __attribute__((ext_vector_type(8)))  __bf16 v8bf;
typedef __attribute__((ext_vector_type(8)))  float  v8f;

#define B_    16
#define L_    512
#define FEAT_ 512
#define N_    4096
#define LLAB_ 32
#define E_    300
#define EPAD_ 320
#define KF_   4
#define KTOT_ 1280   // 4 * 320 (zero-padded K dim for conv GEMM)

// ---- WMMA helpers -----------------------------------------------------------
// A-operand (16x32 bf16): lane<16 holds row m=lane, K = {kb..kb+15}
// with kb = 0; lanes 16-31 hold same rows with kb = 16.  Source must be row-major
// with K contiguous.
__device__ __forceinline__ v16bf load_a16(const bf16* rowp, int lane) {
  int kb = (lane < 16) ? 0 : 16;
  v8bf lo = *(const v8bf*)(rowp + kb);
  v8bf hi = *(const v8bf*)(rowp + kb + 8);
  v16bf r;
#pragma unroll
  for (int i = 0; i < 8; ++i) { r[i] = lo[i]; r[i + 8] = hi[i]; }
  return r;
}
// B-operand (32x16 bf16): lane<16 holds col n=lane, K=0..15; lanes 16-31 hold
// K=16..31.  We feed it from a matrix stored N-major with K contiguous (i.e.
// B-transposed), so each lane does one contiguous 32-byte load.
__device__ __forceinline__ v16bf load_b16(const bf16* rowp, int lane) {
  int kb = (lane < 16) ? 0 : 16;
  return *(const v16bf*)(rowp + kb);
}
__device__ __forceinline__ v8f wmma_bf16(v16bf a, v16bf b, v8f c) {
  return __builtin_amdgcn_wmma_f32_16x16x32_bf16(false, a, false, b,
                                                 (short)0, c, false, false);
}

// ---- prep kernels -----------------------------------------------------------
__global__ void __launch_bounds__(256) prep_x_kernel(const float* __restrict__ x,
                                                     bf16* __restrict__ xb,
                                                     bf16* __restrict__ xtb) {
  int idx = blockIdx.x * 256 + threadIdx.x;
  if (idx >= B_ * L_ * FEAT_) return;
  int b   = idx >> 18;            // 512*512 = 2^18
  int rem = idx & 262143;
  int l   = rem >> 9;
  int f   = rem & 511;
  float v = x[idx];
  xb[idx] = (bf16)v;                         // [b][l][f]
  xtb[(b << 18) + (f << 9) + l] = (bf16)v;   // [b][f][l]
}

__global__ void __launch_bounds__(256) prep_w_kernel(const float* __restrict__ w,
                                                     bf16* __restrict__ wb) {
  int idx = blockIdx.x * 256 + threadIdx.x;
  if (idx >= FEAT_ * KTOT_) return;
  int f = idx / KTOT_;
  int r = idx - f * KTOT_;
  int k = r / EPAD_;
  int e = r - k * EPAD_;
  float v = (e < E_) ? w[(f * E_ + e) * KF_ + k] : 0.0f;  // conv_w (OIH)
  wb[idx] = (bf16)v;   // wb[f][k*320 + e]
}

__global__ void __launch_bounds__(256) prep_lw_kernel(const float* __restrict__ w,
                                                      bf16* __restrict__ wb) {
  int idx = blockIdx.x * 256 + threadIdx.x;
  if (idx >= FEAT_ * FEAT_) return;
  wb[idx] = (bf16)w[idx];
}

// ---- stage 1: conv1d + relu + global maxpool --------------------------------
// One workgroup per label n.  Implicit-GEMM: A[t][k*320+e] = label[t+k][e],
// B^T[f][k*320+e] = conv_w[f][e][k].  Output 32(t,pad) x 512(f); max over
// valid t (0..28), then bias + relu (monotone => commutes with max).
__global__ void __launch_bounds__(256) conv_pool_kernel(
    const float* __restrict__ label_reps, const bf16* __restrict__ wb,
    const float* __restrict__ conv_b, bf16* __restrict__ pool) {
  __shared__ bf16 lab[36 * EPAD_];   // 32 label rows + 4 zero rows for overhang
  int n   = blockIdx.x;
  int tid = threadIdx.x;
  const float* src = label_reps + n * (LLAB_ * E_);
  for (int i = tid; i < 36 * EPAD_; i += 256) {
    int r = i / EPAD_;
    int e = i - r * EPAD_;
    float v = (r < LLAB_ && e < E_) ? src[r * E_ + e] : 0.0f;
    lab[i] = (bf16)v;
  }
  __syncthreads();

  int lane = tid & 31, wave = tid >> 5;
  int m = lane & 15;
  v8f acc[2][4] = {};
  for (int ki = 0; ki < 40; ++ki) {      // 40 K-chunks of 32
    int kf = ki / 10;                    // filter tap (10 chunks per tap)
    int e0 = (ki - kf * 10) * 32;
    v16bf a0 = load_a16(&lab[(m + kf) * EPAD_ + e0], lane);        // t = 0..15
    v16bf a1 = load_a16(&lab[(16 + m + kf) * EPAD_ + e0], lane);   // t = 16..31
#pragma unroll
    for (int j = 0; j < 4; ++j) {
      int f0 = (wave * 4 + j) * 16;
      v16bf bfrag = load_b16(wb + (f0 + m) * KTOT_ + ki * 32, lane);
      acc[0][j] = wmma_bf16(a0, bfrag, acc[0][j]);
      acc[1][j] = wmma_bf16(a1, bfrag, acc[1][j]);
    }
  }
  // Column-wise max over t = 0..28.  C layout: lane<16 elem i -> row i,
  // lane>=16 elem i -> row 8+i, col = lane&15.
#pragma unroll
  for (int j = 0; j < 4; ++j) {
    float mx = -3.0e38f;
#pragma unroll
    for (int i = 0; i < 8; ++i) mx = fmaxf(mx, acc[0][j][i]);
#pragma unroll
    for (int i = 0; i < 8; ++i) {
      int row = (lane < 16) ? i : (8 + i);   // t = 16 + row; valid t <= 28
      if (row <= 12) mx = fmaxf(mx, acc[1][j][i]);
    }
    mx = fmaxf(mx, __shfl_xor(mx, 16, 32));  // combine row halves (same col)
    if (lane < 16) {
      int f = (wave * 4 + j) * 16 + lane;
      pool[n * FEAT_ + f] = (bf16)fmaxf(mx + conv_b[f], 0.0f);
    }
  }
}

// ---- stage 2: linear (pool @ lin_w^T + lin_b) -> bf16 label reps ------------
__global__ void __launch_bounds__(256) linear_kernel(
    const bf16* __restrict__ pool, const bf16* __restrict__ lwb,
    const float* __restrict__ lin_b, bf16* __restrict__ lr) {
  int n0 = blockIdx.x * 16;
  int tid = threadIdx.x;
  int lane = tid & 31, wave = tid >> 5;
  int m = lane & 15;
  v8f acc[4] = {};
  const bf16* abase = pool + (n0 + m) * FEAT_;
  for (int kk = 0; kk < FEAT_; kk += 32) {
    v16bf a = load_a16(abase + kk, lane);
#pragma unroll
    for (int j = 0; j < 4; ++j) {
      int f0 = (wave * 4 + j) * 16;
      v16bf bfrag = load_b16(lwb + (f0 + m) * FEAT_ + kk, lane);  // lin_w[f'][f]
      acc[j] = wmma_bf16(a, bfrag, acc[j]);
    }
  }
#pragma unroll
  for (int j = 0; j < 4; ++j) {
    int col = (wave * 4 + j) * 16 + m;
    float bias = lin_b[col];
#pragma unroll
    for (int i = 0; i < 8; ++i) {
      int row = (lane < 16) ? i : (8 + i);
      lr[(n0 + row) * FEAT_ + col] = (bf16)(acc[j][i] + bias);
    }
  }
}

// ---- stage 3: fused attention (logits -> softmax -> weighted sum) -----------
// One workgroup per (batch b, 16 labels).  att tile never touches HBM.
__global__ void __launch_bounds__(256) attention_kernel(
    const bf16* __restrict__ xb, const bf16* __restrict__ xtb,
    const bf16* __restrict__ lr, float* __restrict__ out) {
  __shared__ float S[16 * 512];      // logits / exp, f32
  __shared__ bf16  P[16 * 512];      // softmax probs, bf16
  __shared__ float red[16][16];
  __shared__ float rstat[16];
  int blk = blockIdx.x;
  int b   = blk >> 8;                // 256 n-tiles per batch
  int n0  = (blk & 255) * 16;
  int tid = threadIdx.x;
  int lane = tid & 31, wave = tid >> 5;
  int m = lane & 15;
  const bf16* xbb  = xb  + ((size_t)b << 18);
  const bf16* xtbb = xtb + ((size_t)b << 18);

  { // GEMM1: S[n][l] = sum_f lr[n][f] * x[l][f]
    v8f acc[4] = {};
    const bf16* abase = lr + (n0 + m) * FEAT_;
    for (int kk = 0; kk < FEAT_; kk += 32) {
      v16bf a = load_a16(abase + kk, lane);
#pragma unroll
      for (int j = 0; j < 4; ++j) {
        int l0 = (wave * 4 + j) * 16;
        v16bf bfrag = load_b16(xbb + (l0 + m) * FEAT_ + kk, lane);
        acc[j] = wmma_bf16(a, bfrag, acc[j]);
      }
    }
#pragma unroll
    for (int j = 0; j < 4; ++j) {
      int col = (wave * 4 + j) * 16 + m;
#pragma unroll
      for (int i = 0; i < 8; ++i) {
        int row = (lane < 16) ? i : (8 + i);
        S[row * 512 + col] = acc[j][i];
      }
    }
  }
  __syncthreads();

  { // row softmax over L = 512 (16 threads per row)
    int r = tid >> 4, c = tid & 15;
    float* Sr = S + r * 512;
    float pmax = -3.0e38f;
    for (int jj = c; jj < 512; jj += 16) pmax = fmaxf(pmax, Sr[jj]);
    red[r][c] = pmax;
    __syncthreads();
    if (c == 0) {
      float mx = red[r][0];
#pragma unroll
      for (int i2 = 1; i2 < 16; ++i2) mx = fmaxf(mx, red[r][i2]);
      rstat[r] = mx;
    }
    __syncthreads();
    float mx = rstat[r];
    float psum = 0.0f;
    for (int jj = c; jj < 512; jj += 16) {
      float e = __expf(Sr[jj] - mx);
      Sr[jj] = e;
      psum += e;
    }
    red[r][c] = psum;
    __syncthreads();
    if (c == 0) {
      float s = 0.0f;
#pragma unroll
      for (int i2 = 0; i2 < 16; ++i2) s += red[r][i2];
      rstat[r] = s;
    }
    __syncthreads();
    float inv = 1.0f / rstat[r];
    bf16* Pr = P + r * 512;
    for (int jj = c; jj < 512; jj += 16) Pr[jj] = (bf16)(Sr[jj] * inv);
  }
  __syncthreads();

  { // GEMM2: out[n][f] = sum_l P[n][l] * x[l][f]   (B-op from x^T)
    v8f acc[4] = {};
    const bf16* abase = P + m * 512;
    for (int ll = 0; ll < 512; ll += 32) {
      v16bf a = load_a16(abase + ll, lane);
#pragma unroll
      for (int j = 0; j < 4; ++j) {
        int f0 = (wave * 4 + j) * 16;
        v16bf bfrag = load_b16(xtbb + (f0 + m) * 512 + ll, lane);
        acc[j] = wmma_bf16(a, bfrag, acc[j]);
      }
    }
    float* outp = out + ((size_t)b * N_ + n0) * FEAT_;
#pragma unroll
    for (int j = 0; j < 4; ++j) {
      int col = (wave * 4 + j) * 16 + m;
#pragma unroll
      for (int i = 0; i < 8; ++i) {
        int row = (lane < 16) ? i : (8 + i);
        outp[row * FEAT_ + col] = acc[j][i];
      }
    }
  }
}

// ---- launcher ---------------------------------------------------------------
extern "C" void kernel_launch(void* const* d_in, const int* in_sizes, int n_in,
                              void* d_out, int out_size, void* d_ws, size_t ws_size,
                              hipStream_t stream) {
  (void)in_sizes; (void)n_in; (void)out_size; (void)ws_size;
  const float* x          = (const float*)d_in[0];
  const float* label_reps = (const float*)d_in[1];
  const float* conv_w     = (const float*)d_in[2];
  const float* conv_b     = (const float*)d_in[3];
  const float* lin_w      = (const float*)d_in[4];
  const float* lin_b      = (const float*)d_in[5];
  float* out = (float*)d_out;

  char* ws = (char*)d_ws;               // ~27 MB total
  bf16* x_bf    = (bf16*)(ws + 0);                 //  8 MB  [b][l][f]
  bf16* xT_bf   = (bf16*)(ws + 8388608);           //  8 MB  [b][f][l]
  bf16* w_bf    = (bf16*)(ws + 16777216);          // 1.25MB [f][k*320+e]
  bf16* lw_bf   = (bf16*)(ws + 18087936);          // 0.5 MB [f'][f]
  bf16* pool_bf = (bf16*)(ws + 18612224);          //  4 MB  [n][f]
  bf16* lr_bf   = (bf16*)(ws + 22806528);          //  4 MB  [n][f']

  prep_x_kernel <<<16384, 256, 0, stream>>>(x, x_bf, xT_bf);
  prep_w_kernel <<<2560,  256, 0, stream>>>(conv_w, w_bf);
  prep_lw_kernel<<<1024,  256, 0, stream>>>(lin_w, lw_bf);
  conv_pool_kernel<<<N_,        256, 0, stream>>>(label_reps, w_bf, conv_b, pool_bf);
  linear_kernel  <<<N_ / 16,    256, 0, stream>>>(pool_bf, lw_bf, lin_b, lr_bf);
  attention_kernel<<<B_ * (N_ / 16), 256, 0, stream>>>(x_bf, xT_bf, lr_bf, out);
}
